// SetGNN_38800734552597
// MI455X (gfx1250) — matrix operand order repaired and checked by reading, not compiled, not run on hardware
//
#include <hip/hip_runtime.h>
#include <hip/hip_bf16.h>
#include <stdint.h>

// ---------------------------------------------------------------------------
// Types for CDNA5 WMMA (wave32): v_wmma_f32_16x16x32_bf16
// ---------------------------------------------------------------------------
typedef __bf16  v16bf __attribute__((ext_vector_type(16)));
typedef float   v8f   __attribute__((ext_vector_type(8)));

union alignas(32) Frag {
    uint4 q[2];
    v16bf v;
};

__device__ __forceinline__ unsigned short f32_to_bf16(float f) {
    unsigned int u = __builtin_bit_cast(unsigned int, f);
    unsigned int r = u + 0x7FFFu + ((u >> 16) & 1u);   // round-to-nearest-even
    return (unsigned short)(r >> 16);
}

#define N_NODES  100000
#define N_HEDGES 20000
#define NNZ_CNT  500000
#define DIM      256
#define MT       64          // rows per workgroup tile
#define LDSTR    264         // bf16 elements per LDS row (256 + 8 pad)
#define LN_EPS   1e-5f

// ---------------------------------------------------------------------------
// Weight prep: fp32 W[k][n] (256x256) -> bf16 WT[n][k] (transposed)
// ---------------------------------------------------------------------------
__global__ void prep_w_kernel(const float* __restrict__ W, unsigned short* __restrict__ WT) {
    int idx = blockIdx.x * 256 + threadIdx.x;     // 256 blocks x 256 thr = 65536
    int k = idx >> 8, n = idx & 255;
    WT[n * 256 + k] = f32_to_bf16(W[k * 256 + n]);
}

// ---------------------------------------------------------------------------
// dst-min reduction (hedge index remap)
// ---------------------------------------------------------------------------
__global__ void init_min_kernel(int* dmin) {
    if (blockIdx.x == 0 && threadIdx.x == 0) *dmin = 0x7fffffff;
}

__global__ void min_kernel(const int* __restrict__ d, int n, int* dmin) {
    int i = blockIdx.x * blockDim.x + threadIdx.x;
    int v = (i < n) ? d[i] : 0x7fffffff;
    #pragma unroll
    for (int m = 16; m > 0; m >>= 1) v = min(v, __shfl_xor(v, m));
    if ((threadIdx.x & 31) == 0) atomicMin(dmin, v);
}

// ---------------------------------------------------------------------------
// Zero a float buffer (float4 granularity)
// ---------------------------------------------------------------------------
__global__ void zero_kernel(float4* __restrict__ p, int n4) {
    int i = blockIdx.x * blockDim.x + threadIdx.x;
    if (i < n4) p[i] = make_float4(0.f, 0.f, 0.f, 0.f);
}

// ---------------------------------------------------------------------------
// Scatter: out[sIdx[e]] += norm[e] * in[gIdx[e]]   (one wave per incidence)
// ---------------------------------------------------------------------------
__global__ __launch_bounds__(256)
void scatter_kernel(const float* __restrict__ in, float* __restrict__ out,
                    const int* __restrict__ gIdx, const int* __restrict__ sIdx,
                    const float* __restrict__ norm, int nnz,
                    const int* __restrict__ dmin, int subG, int subS) {
    int e = blockIdx.x * 8 + (threadIdx.x >> 5);
    if (e >= nnz) return;
    int lane = threadIdx.x & 31;
    int dm = *dmin;
    int g = gIdx[e] - (subG ? dm : 0);
    int s = sIdx[e] - (subS ? dm : 0);
    float w = norm[e];
    const float4* ip = (const float4*)(in + (size_t)g * DIM + lane * 8);
    float4 a = ip[0];
    float4 b = ip[1];
    float* op = out + (size_t)s * DIM + lane * 8;
    atomicAdd(op + 0, w * a.x);
    atomicAdd(op + 1, w * a.y);
    atomicAdd(op + 2, w * a.z);
    atomicAdd(op + 3, w * a.w);
    atomicAdd(op + 4, w * b.x);
    atomicAdd(op + 5, w * b.y);
    atomicAdd(op + 6, w * b.z);
    atomicAdd(op + 7, w * b.w);
}

// ---------------------------------------------------------------------------
// Fused MLP (LN -> GEMM1 +b1,relu -> LN -> GEMM2 +b2,relu), 64-row tiles.
// Wave tiling: 8 waves = 2 row halves (32 rows) x 4 col quarters (64 cols);
// each wave: 2 row sub-blocks x 4 col tiles -> each B fragment feeds 2 WMMAs.
// Safe in-place (in == out).
// ---------------------------------------------------------------------------
__global__ __launch_bounds__(256)
void mlp_kernel(const float* in, float* out, int M,
                const float* __restrict__ ln0g, const float* __restrict__ ln0b,
                const unsigned short* __restrict__ W1T, const float* __restrict__ b1,
                const float* __restrict__ ln1g, const float* __restrict__ ln1b,
                const unsigned short* __restrict__ W2T, const float* __restrict__ b2) {
    __shared__ unsigned short lds_x[MT * LDSTR];   // input LN'd bf16, later h bf16
    __shared__ float lds_sum[MT][4];
    __shared__ float lds_sq[MT][4];
    __shared__ float lds_mu[MT];
    __shared__ float lds_rs[MT];

    const int tid = threadIdx.x;
    const int m0  = blockIdx.x * MT;

    // ---------------- Phase A: input LayerNorm -> bf16 in LDS -------------
    {
        int r = tid >> 2;           // row within tile, 0..63
        int p = tid & 3;            // 64-col chunk
        long gr = (long)m0 + r;
        float s = 0.f, ss = 0.f;
        const float4* xin = (const float4*)(in + gr * DIM + p * 64);
        if (gr < M) {
            #pragma unroll
            for (int i = 0; i < 16; ++i) {
                float4 v = xin[i];
                s  += v.x + v.y + v.z + v.w;
                ss += v.x * v.x + v.y * v.y + v.z * v.z + v.w * v.w;
            }
        }
        s  += __shfl_xor(s, 1);  s  += __shfl_xor(s, 2);
        ss += __shfl_xor(ss, 1); ss += __shfl_xor(ss, 2);
        float mu  = s * (1.f / 256.f);
        float var = ss * (1.f / 256.f) - mu * mu;
        float rs  = rsqrtf(var + LN_EPS);

        unsigned int* lrow = (unsigned int*)&lds_x[r * LDSTR + p * 64];
        if (gr < M) {
            const float4* gg = (const float4*)(ln0g + p * 64);
            const float4* bb = (const float4*)(ln0b + p * 64);
            #pragma unroll
            for (int i = 0; i < 16; ++i) {
                float4 v = xin[i], g4 = gg[i], b4 = bb[i];
                float y0 = (v.x - mu) * rs * g4.x + b4.x;
                float y1 = (v.y - mu) * rs * g4.y + b4.y;
                float y2 = (v.z - mu) * rs * g4.z + b4.z;
                float y3 = (v.w - mu) * rs * g4.w + b4.w;
                lrow[i * 2 + 0] = (unsigned)f32_to_bf16(y0) | ((unsigned)f32_to_bf16(y1) << 16);
                lrow[i * 2 + 1] = (unsigned)f32_to_bf16(y2) | ((unsigned)f32_to_bf16(y3) << 16);
            }
        } else {
            #pragma unroll
            for (int i = 0; i < 32; ++i) lrow[i] = 0u;
        }
    }
    __syncthreads();

    // ------- Wave tiling: 2 row halves (rw) x 4 col quarters (cg) ---------
    const int w    = tid >> 5;
    const int lane = tid & 31;
    const int rw   = w >> 2;              // 0..1 : rows rw*32 .. rw*32+31
    const int cg   = w & 3;               // 0..3 : cols cg*64 .. cg*64+63
    const int lm   = lane & 15;
    const int hsel = (lane >> 4);         // 0: lanes 0-15, 1: lanes 16-31

    // per-col-tile, per-lane column parameters
    float b1v[4], g1v[4], bb1v[4], b2v[4];
    #pragma unroll
    for (int t = 0; t < 4; ++t) {
        int n = cg * 64 + t * 16 + lm;
        b1v[t] = b1[n]; g1v[t] = ln1g[n]; bb1v[t] = ln1b[n]; b2v[t] = b2[n];
    }

    const v8f vzero = {0.f, 0.f, 0.f, 0.f, 0.f, 0.f, 0.f, 0.f};
    v8f acc[2][4];
    #pragma unroll
    for (int sb = 0; sb < 2; ++sb)
        #pragma unroll
        for (int t = 0; t < 4; ++t) acc[sb][t] = vzero;

    const int arow0 = rw * 32 + lm;       // row sub-block 0
    const int arow1 = rw * 32 + 16 + lm;  // row sub-block 1
    const int aoff  = hsel * 8;           // A: K sub-offset (elements)
    const int boff  = hsel * 8;           // B: K sub-offset (elements)

    // ---------------- GEMM1: lds_x(bf16) @ W1T --------------------------
    #pragma unroll
    for (int kb = 0; kb < 8; ++kb) {
        Frag a0, a1, bfr[4];
        a0.q[0] = *(const uint4*)&lds_x[arow0 * LDSTR + kb * 32 + aoff];
        a0.q[1] = *(const uint4*)&lds_x[arow0 * LDSTR + kb * 32 + 16 + aoff];
        a1.q[0] = *(const uint4*)&lds_x[arow1 * LDSTR + kb * 32 + aoff];
        a1.q[1] = *(const uint4*)&lds_x[arow1 * LDSTR + kb * 32 + 16 + aoff];
        #pragma unroll
        for (int t = 0; t < 4; ++t) {
            const uint4* bp = (const uint4*)(W1T + (size_t)(cg * 64 + t * 16 + lm) * 256
                                             + kb * 32 + boff);
            bfr[t].q[0] = bp[0];
            bfr[t].q[1] = bp[2];
        }
        #pragma unroll
        for (int t = 0; t < 4; ++t) {
            acc[0][t] = __builtin_amdgcn_wmma_f32_16x16x32_bf16(
                false, a0.v, false, bfr[t].v, (short)0, acc[0][t], false, false);
            acc[1][t] = __builtin_amdgcn_wmma_f32_16x16x32_bf16(
                false, a1.v, false, bfr[t].v, (short)0, acc[1][t], false, false);
        }
    }
    __syncthreads();   // all reads of lds_x done before overwrite with h

    // bias + relu, then LN1 stats (per-row over post-relu h)
    {
        #pragma unroll
        for (int sb = 0; sb < 2; ++sb) {
            float s[8], ss[8];
            #pragma unroll
            for (int j = 0; j < 8; ++j) { s[j] = 0.f; ss[j] = 0.f; }
            #pragma unroll
            for (int t = 0; t < 4; ++t) {
                #pragma unroll
                for (int j = 0; j < 8; ++j) {
                    float v = acc[sb][t][j] + b1v[t];
                    v = fmaxf(v, 0.f);
                    acc[sb][t][j] = v;
                    s[j]  += v;
                    ss[j] += v * v;
                }
            }
            #pragma unroll
            for (int j = 0; j < 8; ++j) {
                float a0 = s[j], a1 = ss[j];
                a0 += __shfl_xor(a0, 1); a1 += __shfl_xor(a1, 1);
                a0 += __shfl_xor(a0, 2); a1 += __shfl_xor(a1, 2);
                a0 += __shfl_xor(a0, 4); a1 += __shfl_xor(a1, 4);
                a0 += __shfl_xor(a0, 8); a1 += __shfl_xor(a1, 8);
                if (lm == 0) {
                    int m = rw * 32 + sb * 16 + j + (hsel << 3);
                    lds_sum[m][cg] = a0;
                    lds_sq[m][cg]  = a1;
                }
            }
        }
    }
    __syncthreads();
    if (tid < MT) {
        float tot = lds_sum[tid][0] + lds_sum[tid][1] + lds_sum[tid][2] + lds_sum[tid][3];
        float sq  = lds_sq[tid][0] + lds_sq[tid][1] + lds_sq[tid][2] + lds_sq[tid][3];
        float mu  = tot * (1.f / 256.f);
        float q   = sq * (1.f / 256.f) - mu * mu;
        lds_mu[tid] = mu;
        lds_rs[tid] = rsqrtf(q + LN_EPS);
    }
    __syncthreads();

    // normalize h and store bf16 back into lds_x
    #pragma unroll
    for (int sb = 0; sb < 2; ++sb) {
        #pragma unroll
        for (int t = 0; t < 4; ++t) {
            #pragma unroll
            for (int j = 0; j < 8; ++j) {
                int m = rw * 32 + sb * 16 + j + (hsel << 3);
                float hv = (acc[sb][t][j] - lds_mu[m]) * lds_rs[m] * g1v[t] + bb1v[t];
                lds_x[m * LDSTR + cg * 64 + t * 16 + lm] = f32_to_bf16(hv);
            }
            acc[sb][t] = vzero;
        }
    }
    __syncthreads();

    // ---------------- GEMM2: h(bf16) @ W2T ------------------------------
    #pragma unroll
    for (int kb = 0; kb < 8; ++kb) {
        Frag a0, a1, bfr[4];
        a0.q[0] = *(const uint4*)&lds_x[arow0 * LDSTR + kb * 32 + aoff];
        a0.q[1] = *(const uint4*)&lds_x[arow0 * LDSTR + kb * 32 + 16 + aoff];
        a1.q[0] = *(const uint4*)&lds_x[arow1 * LDSTR + kb * 32 + aoff];
        a1.q[1] = *(const uint4*)&lds_x[arow1 * LDSTR + kb * 32 + 16 + aoff];
        #pragma unroll
        for (int t = 0; t < 4; ++t) {
            const uint4* bp = (const uint4*)(W2T + (size_t)(cg * 64 + t * 16 + lm) * 256
                                             + kb * 32 + boff);
            bfr[t].q[0] = bp[0];
            bfr[t].q[1] = bp[2];
        }
        #pragma unroll
        for (int t = 0; t < 4; ++t) {
            acc[0][t] = __builtin_amdgcn_wmma_f32_16x16x32_bf16(
                false, a0.v, false, bfr[t].v, (short)0, acc[0][t], false, false);
            acc[1][t] = __builtin_amdgcn_wmma_f32_16x16x32_bf16(
                false, a1.v, false, bfr[t].v, (short)0, acc[1][t], false, false);
        }
    }

    // bias + relu + guarded store
    #pragma unroll
    for (int sb = 0; sb < 2; ++sb) {
        #pragma unroll
        for (int t = 0; t < 4; ++t) {
            #pragma unroll
            for (int j = 0; j < 8; ++j) {
                long gr = (long)m0 + rw * 32 + sb * 16 + j + (hsel << 3);
                if (gr < M) {
                    float o = fmaxf(acc[sb][t][j] + b2v[t], 0.f);
                    out[gr * DIM + cg * 64 + t * 16 + lm] = o;
                }
            }
        }
    }
}

// ---------------------------------------------------------------------------
// Host-side launcher
// ---------------------------------------------------------------------------
static void launch_mlp(const float* in, float* out, int M, int mlpIdx,
                       void* const* d_in, const unsigned short* wts, hipStream_t stream) {
    int base = 3 + 8 * mlpIdx;   // leaves: ln0_g, ln0_b, W1, b1, ln1_g, ln1_b, W2, b2
    const float* g0  = (const float*)d_in[base + 0];
    const float* b0  = (const float*)d_in[base + 1];
    const float* b1  = (const float*)d_in[base + 3];
    const float* g1  = (const float*)d_in[base + 4];
    const float* bb1 = (const float*)d_in[base + 5];
    const float* b2  = (const float*)d_in[base + 7];
    const unsigned short* w1t = wts + (size_t)(2 * mlpIdx) * 65536;
    const unsigned short* w2t = wts + (size_t)(2 * mlpIdx + 1) * 65536;
    int grid = (M + MT - 1) / MT;
    mlp_kernel<<<grid, 256, 0, stream>>>(in, out, M, g0, b0, w1t, b1, g1, bb1, w2t, b2);
}

extern "C" void kernel_launch(void* const* d_in, const int* in_sizes, int n_in,
                              void* d_out, int out_size, void* d_ws, size_t ws_size,
                              hipStream_t stream) {
    const float* x    = (const float*)d_in[0];
    const int*   ei   = (const int*)d_in[1];     // [2, NNZ]: src then dst
    const float* norm = (const float*)d_in[2];
    const int* src = ei;
    const int* dst = ei + NNZ_CNT;

    // workspace layout
    char* ws = (char*)d_ws;
    unsigned short* wts = (unsigned short*)ws;                    // 16 x 65536 bf16 = 2 MB
    int*   dmin = (int*)(ws + (size_t)16 * 65536 * 2);
    float* bufA = (float*)(ws + (size_t)16 * 65536 * 2 + 256);    // 100000 x 256 f32
    float* bufB = bufA + (size_t)N_NODES * DIM;                   // 100000 x 256 f32
    float* outp = (float*)d_out;

    // --- weight prep: bf16 transposed copies of W1/W2 for all 8 MLPs ---
    for (int m = 0; m < 8; ++m) {
        prep_w_kernel<<<256, 256, 0, stream>>>((const float*)d_in[3 + 8 * m + 2],
                                               wts + (size_t)(2 * m) * 65536);
        prep_w_kernel<<<256, 256, 0, stream>>>((const float*)d_in[3 + 8 * m + 6],
                                               wts + (size_t)(2 * m + 1) * 65536);
    }

    // --- hedge index remap: min(dst) ---
    init_min_kernel<<<1, 32, 0, stream>>>(dmin);
    min_kernel<<<(NNZ_CNT + 255) / 256, 256, 0, stream>>>(dst, NNZ_CNT, dmin);

    const int nzE = N_HEDGES * DIM / 4;
    const int nzN = N_NODES * DIM / 4;
    const int sgrid = (NNZ_CNT + 7) / 8;

    // MLP order: V2E0(enc=0,dec=1), V2E1(enc=2,dec=3), E2V0(enc=4,dec=5), E2V1(enc=6,dec=7)
    // ---- Layer 0 ----
    launch_mlp(x, bufA, N_NODES, 0, d_in, wts, stream);                         // enc V2E0
    zero_kernel<<<(nzE + 255) / 256, 256, 0, stream>>>((float4*)bufB, nzE);
    scatter_kernel<<<sgrid, 256, 0, stream>>>(bufA, bufB, src, dst, norm,
                                              NNZ_CNT, dmin, 0, 1);             // nodes -> hedges
    launch_mlp(bufB, bufB, N_HEDGES, 1, d_in, wts, stream);                     // dec V2E0
    launch_mlp(bufB, bufB, N_HEDGES, 4, d_in, wts, stream);                     // enc E2V0
    zero_kernel<<<(nzN + 255) / 256, 256, 0, stream>>>((float4*)bufA, nzN);
    scatter_kernel<<<sgrid, 256, 0, stream>>>(bufB, bufA, dst, src, norm,
                                              NNZ_CNT, dmin, 1, 0);             // hedges -> nodes
    launch_mlp(bufA, bufA, N_NODES, 5, d_in, wts, stream);                      // dec E2V0

    // ---- Layer 1 ----
    launch_mlp(bufA, bufA, N_NODES, 2, d_in, wts, stream);                      // enc V2E1
    zero_kernel<<<(nzE + 255) / 256, 256, 0, stream>>>((float4*)bufB, nzE);
    scatter_kernel<<<sgrid, 256, 0, stream>>>(bufA, bufB, src, dst, norm,
                                              NNZ_CNT, dmin, 0, 1);
    launch_mlp(bufB, bufB, N_HEDGES, 3, d_in, wts, stream);                     // dec V2E1
    launch_mlp(bufB, bufB, N_HEDGES, 6, d_in, wts, stream);                     // enc E2V1
    zero_kernel<<<(nzN + 255) / 256, 256, 0, stream>>>((float4*)bufA, nzN);
    scatter_kernel<<<sgrid, 256, 0, stream>>>(bufB, bufA, dst, src, norm,
                                              NNZ_CNT, dmin, 1, 0);
    launch_mlp(bufA, outp, N_NODES, 7, d_in, wts, stream);                      // dec E2V1 -> d_out
}
